// CausalMultiHeadedAttention_1142461301185
// MI455X (gfx1250) — matrix operand order repaired and checked by reading, not compiled, not run on hardware
//
#include <hip/hip_runtime.h>

#define Bn  2
#define Tn  2048
#define Dn  1024
#define Hn  16
#define DHn 64
#define BTn (Bn * Tn)

typedef __attribute__((ext_vector_type(16))) __bf16 v16bf;
typedef __attribute__((ext_vector_type(8)))  __bf16 v8bf;
typedef __attribute__((ext_vector_type(4)))  __bf16 v4bf;
typedef __attribute__((ext_vector_type(2)))  __bf16 v2bf;
typedef __attribute__((ext_vector_type(8)))  float  v8f;

__device__ __forceinline__ v16bf combine16(v8bf lo, v8bf hi) {
  v16bf r;
#pragma unroll
  for (int i = 0; i < 8; ++i) { r[i] = lo[i]; r[i + 8] = hi[i]; }
  return r;
}

// A fragment: 16x32 bf16, row-major source, row stride ld (elements).
// lane L<16: row M=L, K={koff..koff+7, 16+koff..16+koff+7}, koff=8*(L>=16)
__device__ __forceinline__ v16bf load_a_frag(const __bf16* base, int ld, int lane) {
  const int row  = lane & 15;
  const int koff = (lane >> 4) << 3;        // 0 or 8
  const __bf16* r = base + (size_t)row * ld;
  v8bf lo = *(const v8bf*)(r + koff);
  v8bf hi = *(const v8bf*)(r + 16 + koff);
  return combine16(lo, hi);
}

// B fragment: 32x16 bf16 from an n-major buffer Bt[n][k] (row stride ld).
__device__ __forceinline__ v16bf load_bt_frag(const __bf16* bt, int ld, int lane) {
  const int n    = lane & 15;
  const int koff = (lane >> 4) << 3;        // 0 or 8
  const __bf16* r = bt + (size_t)n * ld;
  v8bf lo = *(const v8bf*)(r + koff);
  v8bf hi = *(const v8bf*)(r + 16 + koff);
  return combine16(lo, hi);
}

__device__ __forceinline__ v8f wmma_bf16(v16bf a, v16bf b, v8f c) {
  return __builtin_amdgcn_wmma_f32_16x16x32_bf16(false, a, false, b, (short)0, c, false, false);
}

// ---------------------------------------------------------------------------
// Kernel 1: fused K/Q/V projection.  One block stages the X tile once and
// produces the 128x64 tile of all three outputs (12 WMMAs / k-step).
// ---------------------------------------------------------------------------
__global__ __launch_bounds__(256) void qkv_proj_kernel(
    const float* __restrict__ X,
    const float* __restrict__ Wk, const float* __restrict__ Wq,
    const float* __restrict__ Wv,
    __bf16* __restrict__ Kb, __bf16* __restrict__ Qb, __bf16* __restrict__ Vb) {
  __shared__ __bf16 Xs[128][40]     __attribute__((aligned(16)));  // [row][k]
  __shared__ __bf16 WsT[3][64][40]  __attribute__((aligned(16)));  // [mat][n][k]

  const int tid  = threadIdx.x;
  const int lane = tid & 31;
  const int wv   = tid >> 5;
  const int row0 = blockIdx.y * 128;
  const int col0 = blockIdx.x * 64;

  const float* Wm[3] = {Wk, Wq, Wv};

  v8f acc[3][4];
#pragma unroll
  for (int mt = 0; mt < 3; ++mt)
#pragma unroll
    for (int cb = 0; cb < 4; ++cb)
#pragma unroll
      for (int i = 0; i < 8; ++i) acc[mt][cb][i] = 0.0f;

  for (int kk = 0; kk < Dn; kk += 32) {
    // X tile 128x32: float4 loads -> packed bf16 (ds_store_b64)
#pragma unroll
    for (int t = 0; t < 4; ++t) {
      int u = tid + t * 256;                // 0..1023, 8 float4 per row
      int r = u >> 3;
      int c = (u & 7) * 4;
      const float4 x4 = *(const float4*)&X[(size_t)(row0 + r) * Dn + kk + c];
      v4bf b4;
      b4[0] = (__bf16)x4.x; b4[1] = (__bf16)x4.y;
      b4[2] = (__bf16)x4.z; b4[3] = (__bf16)x4.w;
      *(v4bf*)&Xs[r][c] = b4;
    }
    // W tiles 32x64 (x3): coalesced row reads, pack 2 k's per ds_store_b32
#pragma unroll
    for (int mt = 0; mt < 3; ++mt)
#pragma unroll
      for (int t = 0; t < 4; ++t) {
        int u  = tid + t * 256;             // 0..1023
        int nn = u & 63;
        int k2 = (u >> 6) << 1;             // even k
        float w0 = Wm[mt][(size_t)(kk + k2)     * Dn + col0 + nn];
        float w1 = Wm[mt][(size_t)(kk + k2 + 1) * Dn + col0 + nn];
        v2bf p; p[0] = (__bf16)w0; p[1] = (__bf16)w1;
        *(v2bf*)&WsT[mt][nn][k2] = p;
      }
    if (kk + 32 < Dn) {
      int pr = tid & 127;
      __builtin_prefetch(&X[(size_t)(row0 + pr) * Dn + kk + 32], 0, 3);
      int pk = tid & 31;
      __builtin_prefetch(&Wk[(size_t)(kk + 32 + pk) * Dn + col0], 0, 3);
      __builtin_prefetch(&Wq[(size_t)(kk + 32 + pk) * Dn + col0], 0, 3);
      __builtin_prefetch(&Wv[(size_t)(kk + 32 + pk) * Dn + col0], 0, 3);
    }
    __syncthreads();
    v16bf a = load_a_frag(&Xs[wv * 16][0], 40, lane);
#pragma unroll
    for (int mt = 0; mt < 3; ++mt)
#pragma unroll
      for (int cb = 0; cb < 4; ++cb) {
        v16bf bb = load_bt_frag(&WsT[mt][cb * 16][0], 40, lane);
        acc[mt][cb] = wmma_bf16(a, bb, acc[mt][cb]);
      }
    __syncthreads();
  }

  __bf16* Cm[3] = {Kb, Qb, Vb};
  const int n  = lane & 15;
  const int mh = (lane >> 4) * 8;
#pragma unroll
  for (int mt = 0; mt < 3; ++mt)
#pragma unroll
    for (int cb = 0; cb < 4; ++cb)
#pragma unroll
      for (int v = 0; v < 8; ++v) {
        int r = row0 + wv * 16 + mh + v;
        Cm[mt][(size_t)r * Dn + col0 + cb * 16 + n] = (__bf16)acc[mt][cb][v];
      }
}

// ---------------------------------------------------------------------------
// Kernel 2: flash attention.  scores[i,j] = (k_i . q_j)/8, causal j<=i,
// softmax over j, O = P @ V.  Block: (b,h, 128-row i-block), 8 waves x 16 rows.
// ---------------------------------------------------------------------------
__global__ __launch_bounds__(256) void attn_kernel(
    const __bf16* __restrict__ Kb, const __bf16* __restrict__ Qb,
    const __bf16* __restrict__ Vb, __bf16* __restrict__ Cx) {
  const int bh    = blockIdx.x;
  const int b     = bh >> 4;
  const int h     = bh & 15;
  const int ibase = blockIdx.y * 128;

  const int tid  = threadIdx.x;
  const int lane = tid & 31;
  const int wv   = tid >> 5;

  const size_t base = (size_t)b * Tn * Dn + (size_t)h * DHn;
  const __bf16* Kp = Kb + base;
  const __bf16* Qp = Qb + base;
  const __bf16* Vp = Vb + base;

  __shared__ __bf16 Vt[64][72]    __attribute__((aligned(16)));  // [d][j_local]
  __shared__ __bf16 Pl[8][16][72] __attribute__((aligned(16)));  // per-wave P strip

  v8f o[4];
  float m[8], l[8];
#pragma unroll
  for (int cb = 0; cb < 4; ++cb)
#pragma unroll
    for (int i = 0; i < 8; ++i) o[cb][i] = 0.0f;
#pragma unroll
  for (int v = 0; v < 8; ++v) { m[v] = -1e30f; l[v] = 0.0f; }

  const int irow = ibase + wv * 16;
  const int n    = lane & 15;
  const int mh   = (lane >> 4) * 8;
  const int njb  = (ibase >> 6) + 2;               // only j-blocks <= diagonal

  // K A-fragments for this wave's 16-row strip are loop-invariant: hoist.
  v16bf ka[2];
#pragma unroll
  for (int kb = 0; kb < 2; ++kb)
    ka[kb] = load_a_frag(Kp + (size_t)irow * Dn + kb * 32, Dn, lane);

  for (int jb = 0; jb < njb; ++jb) {
    const int jbase = jb * 64;

    // prefetch next j-block of Q and V
    if (jb + 1 < njb) {
      int pj = tid & 63;
      __builtin_prefetch(&Qp[(size_t)(jbase + 64 + pj) * Dn], 0, 0);
      __builtin_prefetch(&Vp[(size_t)(jbase + 64 + pj) * Dn], 0, 0);
    }

    // stage V tile transposed: Vt[d][j] = V[jbase+j][d]
    // two j-rows per thread, packed v2bf ds_store_b32
    {
      int j2 = (tid >> 3) << 1;                    // 0..62 even
      int d  = (tid & 7) * 8;
      v8bf r0 = *(const v8bf*)&Vp[(size_t)(jbase + j2)     * Dn + d];
      v8bf r1 = *(const v8bf*)&Vp[(size_t)(jbase + j2 + 1) * Dn + d];
#pragma unroll
      for (int e = 0; e < 8; ++e) {
        v2bf p; p[0] = r0[e]; p[1] = r1[e];
        *(v2bf*)&Vt[d + e][j2] = p;
      }
    }
    __syncthreads();

    // S = K_strip(16x64) @ Q_tile^T(64x64)
    v8f s[4];
#pragma unroll
    for (int cb = 0; cb < 4; ++cb)
#pragma unroll
      for (int i = 0; i < 8; ++i) s[cb][i] = 0.0f;
#pragma unroll
    for (int kb = 0; kb < 2; ++kb) {
#pragma unroll
      for (int cb = 0; cb < 4; ++cb) {
        v16bf bq = load_bt_frag(Qp + (size_t)(jbase + cb * 16) * Dn + kb * 32,
                                Dn, lane);
        s[cb] = wmma_bf16(ka[kb], bq, s[cb]);
      }
    }

    // scale, causal mask, row max (rows live on 16-lane halves)
    float rmax[8];
#pragma unroll
    for (int v = 0; v < 8; ++v) rmax[v] = -1e30f;
#pragma unroll
    for (int cb = 0; cb < 4; ++cb)
#pragma unroll
      for (int v = 0; v < 8; ++v) {
        int j = jbase + cb * 16 + n;
        int i = irow + mh + v;
        float sv = s[cb][v] * 0.125f;
        if (j > i) sv = -1e30f;
        s[cb][v] = sv;
        rmax[v] = fmaxf(rmax[v], sv);
      }
#pragma unroll
    for (int off = 1; off < 16; off <<= 1)
#pragma unroll
      for (int v = 0; v < 8; ++v)
        rmax[v] = fmaxf(rmax[v], __shfl_xor(rmax[v], off, 32));

    float alpha[8], rsum[8];
#pragma unroll
    for (int v = 0; v < 8; ++v) {
      float mn = fmaxf(m[v], rmax[v]);
      alpha[v] = __expf(m[v] - mn);
      m[v] = mn;
      rsum[v] = 0.0f;
    }
#pragma unroll
    for (int cb = 0; cb < 4; ++cb)
#pragma unroll
      for (int v = 0; v < 8; ++v) {
        float p = __expf(s[cb][v] - m[v]);
        s[cb][v] = p;
        rsum[v] += p;
      }
#pragma unroll
    for (int off = 1; off < 16; off <<= 1)
#pragma unroll
      for (int v = 0; v < 8; ++v)
        rsum[v] += __shfl_xor(rsum[v], off, 32);
#pragma unroll
    for (int v = 0; v < 8; ++v) l[v] = l[v] * alpha[v] + rsum[v];
#pragma unroll
    for (int cb = 0; cb < 4; ++cb)
#pragma unroll
      for (int v = 0; v < 8; ++v) o[cb][v] *= alpha[v];

    // stash P as bf16 into this wave's private LDS strip (A-frag source)
#pragma unroll
    for (int cb = 0; cb < 4; ++cb)
#pragma unroll
      for (int v = 0; v < 8; ++v)
        Pl[wv][mh + v][cb * 16 + n] = (__bf16)s[cb][v];

    // O += P(16x64) @ V(64x64)
#pragma unroll
    for (int kk = 0; kk < 64; kk += 32) {
      v16bf a = load_a_frag(&Pl[wv][0][kk], 72, lane);
#pragma unroll
      for (int cb = 0; cb < 4; ++cb) {
        v16bf bvv = load_bt_frag(&Vt[cb * 16][kk], 72, lane);
        o[cb] = wmma_bf16(a, bvv, o[cb]);
      }
    }
    __syncthreads();
  }

  // epilogue: divide by l, store ctx (bf16) in [b*T+t, h*64+d] layout
#pragma unroll
  for (int cb = 0; cb < 4; ++cb)
#pragma unroll
    for (int v = 0; v < 8; ++v) {
      float val = o[cb][v] / l[v];
      int t = irow + mh + v;
      Cx[(size_t)b * Tn * Dn + (size_t)t * Dn + h * DHn + cb * 16 + n] = (__bf16)val;
    }
}

// ---------------------------------------------------------------------------
// Kernel 3: output projection. Out(fp32) = ctx(bf16) @ W_out(fp32) + b_out.
// ctx tile is a straight bf16 copy -> staged with CDNA5 async global->LDS.
// ---------------------------------------------------------------------------
__global__ __launch_bounds__(256) void out_proj_kernel(
    const __bf16* __restrict__ A, const float* __restrict__ W,
    const float* __restrict__ bias, float* __restrict__ Out) {
  __shared__ __bf16 Xs[128][40] __attribute__((aligned(16)));
  __shared__ __bf16 WsT[64][40] __attribute__((aligned(16)));

  const int tid  = threadIdx.x;
  const int lane = tid & 31;
  const int wv   = tid >> 5;
  const int row0 = blockIdx.y * 128;
  const int col0 = blockIdx.x * 64;

  v8f acc[4];
#pragma unroll
  for (int cb = 0; cb < 4; ++cb)
#pragma unroll
    for (int i = 0; i < 8; ++i) acc[cb][i] = 0.0f;

  for (int kk = 0; kk < Dn; kk += 32) {
    // ctx tile 128x32 bf16: async DMA global->LDS, 16B per lane per issue
#pragma unroll
    for (int t = 0; t < 2; ++t) {
      int u = tid + t * 256;                       // 0..511
      int r = u >> 2;
      int c = (u & 3) * 8;
      unsigned lds_addr =
          (unsigned)(unsigned long long)(const void*)&Xs[r][c];
      unsigned long long gaddr =
          (unsigned long long)(const void*)&A[(size_t)(row0 + r) * Dn + kk + c];
      asm volatile("global_load_async_to_lds_b128 %0, %1, off"
                   :: "v"(lds_addr), "v"(gaddr) : "memory");
    }
    // W tile 32x64: coalesced reads, pack 2 k's per ds_store_b32
#pragma unroll
    for (int t = 0; t < 4; ++t) {
      int u  = tid + t * 256;                      // 0..1023
      int nn = u & 63;
      int k2 = (u >> 6) << 1;
      float w0 = W[(size_t)(kk + k2)     * Dn + col0 + nn];
      float w1 = W[(size_t)(kk + k2 + 1) * Dn + col0 + nn];
      v2bf p; p[0] = (__bf16)w0; p[1] = (__bf16)w1;
      *(v2bf*)&WsT[nn][k2] = p;
    }
    if (kk + 32 < Dn) {
      int pr = tid & 127;
      __builtin_prefetch(&A[(size_t)(row0 + pr) * Dn + kk + 32], 0, 3);
      int pk = tid & 31;
      __builtin_prefetch(&W[(size_t)(kk + 32 + pk) * Dn + col0], 0, 3);
    }
    asm volatile("s_wait_asynccnt 0" ::: "memory");
    __syncthreads();

    v16bf a = load_a_frag(&Xs[wv * 16][0], 40, lane);
#pragma unroll
    for (int cb = 0; cb < 4; ++cb) {
      v16bf bb = load_bt_frag(&WsT[cb * 16][0], 40, lane);
      acc[cb] = wmma_bf16(a, bb, acc[cb]);
    }
    __syncthreads();
  }

  const int n  = lane & 15;
  const int mh = (lane >> 4) * 8;
#pragma unroll
  for (int cb = 0; cb < 4; ++cb)
#pragma unroll
    for (int v = 0; v < 8; ++v) {
      int r   = row0 + wv * 16 + mh + v;
      int col = col0 + cb * 16 + n;
      Out[(size_t)r * Dn + col] = acc[cb][v] + bias[col];
    }
}

// ---------------------------------------------------------------------------
extern "C" void kernel_launch(void* const* d_in, const int* in_sizes, int n_in,
                              void* d_out, int out_size, void* d_ws, size_t ws_size,
                              hipStream_t stream) {
  const float* X  = (const float*)d_in[0];
  const float* Wk = (const float*)d_in[1];
  const float* Wq = (const float*)d_in[2];
  const float* Wv = (const float*)d_in[3];
  const float* Wo = (const float*)d_in[4];
  const float* bo = (const float*)d_in[5];
  float* Out = (float*)d_out;

  const size_t elems = (size_t)BTn * Dn;              // 4M elements
  char* ws = (char*)d_ws;
  __bf16* Kb = (__bf16*)(ws);
  __bf16* Qb = (__bf16*)(ws + elems * 2);
  __bf16* Vb = (__bf16*)(ws + elems * 4);
  __bf16* Cx = (__bf16*)(ws + elems * 6);

  qkv_proj_kernel<<<dim3(Dn / 64, BTn / 128), 256, 0, stream>>>(
      X, Wk, Wq, Wv, Kb, Qb, Vb);
  attn_kernel<<<dim3(Bn * Hn, Tn / 128), 256, 0, stream>>>(Kb, Qb, Vb, Cx);
  out_proj_kernel<<<dim3(Dn / 64, BTn / 128), 256, 0, stream>>>(Cx, Wo, bo, Out);
}
